// LidarGAT_20779051778366
// MI455X (gfx1250) — matrix operand order repaired and checked by reading, not compiled, not run on hardware
//
#include <hip/hip_runtime.h>
#include <math.h>

// ---------------------------------------------------------------------------
// CDNA5 (gfx1250) GAT pipeline.
//  - GEMMs: 64x64 block tiles, LDS-staged A/B, 4x v_wmma_f32_16x16x32_bf16
//    per K-step per wave (wave32, WGP LDS).
//  - TDM demo kernel: tensor_load_to_lds via inline asm + s_wait_tensorcnt.
//  - Attention softmax/scatter: edge-parallel kernels with f32 atomics.
// ---------------------------------------------------------------------------

typedef __bf16 bf16_t;
typedef __attribute__((ext_vector_type(16))) __bf16 v16bf;
typedef __attribute__((ext_vector_type(8)))  float  v8f;
typedef unsigned int u32x4 __attribute__((ext_vector_type(4)));
typedef unsigned int u32x8 __attribute__((ext_vector_type(8)));

static __device__ __forceinline__ void atomic_max_f32(float* addr, float v) {
  // IEEE-754 ordering trick: signed-int max for v>=0, unsigned-int min for v<0.
  if (v >= 0.0f) atomicMax((int*)addr, __float_as_int(v));
  else           atomicMin((unsigned int*)addr, __float_as_uint(v));
}

// ---------------- utility fills / conversions ------------------------------

__global__ void k_fill_f32(float* __restrict__ p, float v, size_t n) {
  size_t i = (size_t)blockIdx.x * blockDim.x + threadIdx.x;
  if (i < n) p[i] = v;
}

// fp32 [rows_in, cols] -> bf16 [rows_out, cols], zero-padding extra rows (K pad)
__global__ void k_cvt_rows_pad(const float* __restrict__ W, bf16_t* __restrict__ out,
                               int rows_in, int rows_out, int cols) {
  int i = blockIdx.x * blockDim.x + threadIdx.x;
  int total = rows_out * cols;
  if (i >= total) return;
  int r = i / cols, c = i - r * cols;
  out[i] = (r < rows_in) ? (bf16_t)W[r * cols + c] : (bf16_t)0.0f;
}

// fp32 [n, cin] -> bf16 [n, cpad], zero-padding extra columns
__global__ void k_cvt_cols_pad(const float* __restrict__ X, bf16_t* __restrict__ out,
                               int n, int cin, int cpad) {
  size_t i = (size_t)blockIdx.x * blockDim.x + threadIdx.x;
  size_t total = (size_t)n * cpad;
  if (i >= total) return;
  int c = (int)(i % cpad);
  size_t r = i / cpad;
  out[i] = (c < cin) ? (bf16_t)X[r * cin + c] : (bf16_t)0.0f;
}

// ---------------- TDM demo: DMA a 16x16 f32 tile into LDS ------------------
// Uses the Tensor Data Mover (CDNA5 ISA) via inline asm
// (portable across toolchains). D# group layout per ISA §8.3/8.4:
//  g0: {count=1 | lds_addr | global_addr[31:0] | global_addr[56:32], type=2}
//  g1: {data_size=2(4B) | tensor_dim0/1=16 | tile_dim0/1=16 | stride0=16}
__global__ void k_tdm_copy16x16(const float* __restrict__ src, float* __restrict__ dst) {
  __shared__ float tile[16 * 16];
  u32x4 g0;
  unsigned long long ga = (unsigned long long)(size_t)src;
  g0[0] = 1u;                                        // count=1, user mode
  g0[1] = (unsigned)(size_t)tile;                    // lds_addr (byte offset)
  g0[2] = (unsigned)(ga & 0xffffffffu);              // global_addr[31:0]
  g0[3] = (unsigned)((ga >> 32) & 0x01ffffffu) | (2u << 30);  // addr[56:32] | type=2
  u32x8 g1;
  g1[0] = 2u << 16;                                  // workgroup_mask=0, data_size=4B
  g1[1] = (16u & 0xffffu) << 16;                     // tensor_dim0[15:0]=16
  g1[2] = 0u | (16u << 16);                          // dim0[31:16]=0, tensor_dim1[15:0]=16
  g1[3] = 0u | (16u << 16);                          // dim1[31:16]=0, tile_dim0=16
  g1[4] = 16u;                                       // tile_dim1=16, tile_dim2=0
  g1[5] = 16u;                                       // tensor_dim0_stride[31:0]=16
  g1[6] = 0u;                                        // stride0[47:32], stride1[15:0]
  g1[7] = 0u;                                        // stride1[47:16]
  u32x4 g2 = {0u, 0u, 0u, 0u};
  u32x4 g3 = {0u, 0u, 0u, 0u};
  asm volatile("tensor_load_to_lds %0, %1, %2, %3"
               :: "s"(g0), "s"(g1), "s"(g2), "s"(g3) : "memory");
  asm volatile("s_wait_tensorcnt 0x0" ::: "memory");
  __syncthreads();
  if (threadIdx.x < 256) dst[threadIdx.x] = tile[threadIdx.x];
}

// ---------------- bf16 WMMA GEMM: C[M,N] = A[M,K] * B[K,N] (row-major) -----
// Block = 128 threads (4 waves) computing a 64x64 C tile. Each K-step stages
// A (64x32) and B-transposed (64n x 32k) into LDS with padded pitch, then each
// wave issues 4 WMMAs (one per 16-col subtile) from contiguous LDS fragments.
// Fragment layouts per CDNA5 ISA §7.12.2.
#define TK     32
#define LPITCH 40   // bf16 elems per LDS row (32 used + 8 pad -> conflict-free)

__global__ __launch_bounds__(128)
void k_gemm_bf16(const bf16_t* __restrict__ A, const bf16_t* __restrict__ Bm,
                 float* __restrict__ C, int M, int K, int N) {
  __shared__ bf16_t lA[64 * LPITCH];   // [row][k]
  __shared__ bf16_t lB[64 * LPITCH];   // [n][k]  (transposed)

  const int lane = threadIdx.x;                 // 0..31
  const int wv   = threadIdx.y;                 // 0..3
  const int tid  = wv * 32 + lane;              // 0..127
  const int m    = lane & 15;
  const int kh   = lane >> 4;
  const int row0 = blockIdx.y * 64;
  const int n0   = blockIdx.x * 64;

  v8f acc[4] = {};

  for (int k0 = 0; k0 < K; k0 += TK) {
    // ---- stage A: 64 rows x 32 k (8 bf16 = 16B per thread-slot) ----
#pragma unroll
    for (int pass = 0; pass < 2; ++pass) {
      int p  = pass * 128 + tid;                // 0..255
      int r  = p >> 2;                          // 0..63
      int kc = (p & 3) * 8;                     // 0,8,16,24
      int gr = row0 + r;
      uint4 v = make_uint4(0u, 0u, 0u, 0u);
      if (gr < M) {
        const bf16_t* gp = A + (size_t)gr * K + k0 + kc;
        v = *(const uint4*)gp;
        if (k0 + TK < K) __builtin_prefetch(gp + TK, 0, 0);
      }
      *(uint4*)(&lA[r * LPITCH + kc]) = v;
    }
    // ---- stage B transposed: lB[n][k], n<64, k<32 ----
    {
      int kb = tid >> 4;                        // 0..7
      int ng = (tid & 15) * 4;                  // 0,4,...,60
#pragma unroll
      for (int kk = 0; kk < 4; ++kk) {
        int k  = kb + kk * 8;
        const bf16_t* bp = Bm + (size_t)(k0 + k) * N + n0 + ng;
        bf16_t e0 = (bf16_t)0.0f, e1 = e0, e2 = e0, e3 = e0;
        if (n0 + ng + 3 < N) {                  // N is a multiple of 4 here
          e0 = bp[0]; e1 = bp[1]; e2 = bp[2]; e3 = bp[3];
          if (k0 + TK < K) __builtin_prefetch(bp + (size_t)TK * N, 0, 0);
        }
        lB[(ng + 0) * LPITCH + k] = e0;
        lB[(ng + 1) * LPITCH + k] = e1;
        lB[(ng + 2) * LPITCH + k] = e2;
        lB[(ng + 3) * LPITCH + k] = e3;
      }
    }
    __syncthreads();

    // ---- fragments + 4 WMMAs ----
    v16bf a;
    const bf16_t* ap = &lA[(wv * 16 + m) * LPITCH + kh * 8];
#pragma unroll
    for (int i = 0; i < 8; ++i) { a[i] = ap[i]; a[8 + i] = ap[16 + i]; }
#pragma unroll
    for (int nt = 0; nt < 4; ++nt) {
      const bf16_t* bp = &lB[(nt * 16 + m) * LPITCH + kh * 8];
      v16bf b;
#pragma unroll
      for (int i = 0; i < 8; ++i) { b[i] = bp[i]; b[8 + i] = bp[16 + i]; }
      acc[nt] = __builtin_amdgcn_wmma_f32_16x16x32_bf16(false, a, false, b,
                                                        (short)0, acc[nt], false, false);
    }
    __syncthreads();
  }

  // ---- store C: vgpr r -> row kh*8+r, col = lane&15 ----
#pragma unroll
  for (int nt = 0; nt < 4; ++nt) {
    int col = n0 + nt * 16 + m;
    if (col < N) {
#pragma unroll
      for (int r = 0; r < 8; ++r) {
        int rr = row0 + wv * 16 + kh * 8 + r;
        if (rr < M) C[(size_t)rr * N + col] = acc[nt][r];
      }
    }
  }
}

// ---------------- per-node attention logits --------------------------------
__global__ void k_node_logits(const float* __restrict__ h,
                              const float* __restrict__ a_src,
                              const float* __restrict__ a_dst,
                              float* __restrict__ al_s, float* __restrict__ al_d,
                              int n, int heads, int dout) {
  int idx = blockIdx.x * blockDim.x + threadIdx.x;
  if (idx >= n * heads) return;
  int node = idx / heads, hd = idx - node * heads;
  const float* hp = h + (size_t)node * heads * dout + (size_t)hd * dout;
  const float* as = a_src + (size_t)hd * dout;
  const float* ad = a_dst + (size_t)hd * dout;
  float s = 0.f, d = 0.f;
  for (int j = 0; j < dout; ++j) { float v = hp[j]; s += v * as[j]; d += v * ad[j]; }
  al_s[idx] = s;
  al_d[idx] = d;
}

// ---------------- edge passes (softmax over incoming edges) ----------------
// edge id e < E: src=ei[e], dst=ei[E+e];  e >= E: self loop (e-E, e-E)

__global__ void k_edge_scores(const int* __restrict__ ei, int E, int N,
                              const float* __restrict__ als, const float* __restrict__ ald,
                              float* __restrict__ ebuf, float* __restrict__ emax, int heads) {
  size_t t = (size_t)blockIdx.x * blockDim.x + threadIdx.x;
  size_t total = (size_t)(E + N) * heads;
  if (t >= total) return;
  int hd = (int)(t % heads);
  size_t e = t / heads;
  int src, dst;
  if (e < (size_t)E) { src = ei[e]; dst = ei[(size_t)E + e]; }
  else               { src = dst = (int)(e - (size_t)E); }
  float v = als[(size_t)src * heads + hd] + ald[(size_t)dst * heads + hd];
  v = (v > 0.f) ? v : 0.2f * v;                     // LeakyReLU(0.2)
  ebuf[t] = v;
  atomic_max_f32(&emax[(size_t)dst * heads + hd], v);
}

__global__ void k_edge_expsum(const int* __restrict__ ei, int E, int N,
                              float* __restrict__ ebuf, const float* __restrict__ emax,
                              float* __restrict__ den, int heads) {
  size_t t = (size_t)blockIdx.x * blockDim.x + threadIdx.x;
  size_t total = (size_t)(E + N) * heads;
  if (t >= total) return;
  int hd = (int)(t % heads);
  size_t e = t / heads;
  int dst = (e < (size_t)E) ? ei[(size_t)E + e] : (int)(e - (size_t)E);
  float ex = __expf(ebuf[t] - emax[(size_t)dst * heads + hd]);
  ebuf[t] = ex;
  atomicAdd(&den[(size_t)dst * heads + hd], ex);
}

__global__ void k_edge_alpha(const int* __restrict__ ei, int E, int N,
                             float* __restrict__ ebuf, const float* __restrict__ den,
                             int heads) {
  size_t t = (size_t)blockIdx.x * blockDim.x + threadIdx.x;
  size_t total = (size_t)(E + N) * heads;
  if (t >= total) return;
  int hd = (int)(t % heads);
  size_t e = t / heads;
  int dst = (e < (size_t)E) ? ei[(size_t)E + e] : (int)(e - (size_t)E);
  ebuf[t] = ebuf[t] / den[(size_t)dst * heads + hd];
}

// scatter messages: out[dst,f] += h[src,f] * alpha[e, f/dout]
__global__ void k_edge_aggregate(const int* __restrict__ ei, int E, int N,
                                 const float* __restrict__ h, const float* __restrict__ alpha,
                                 float* __restrict__ out, int heads, int dout) {
  const int F = heads * dout;
  size_t t = (size_t)blockIdx.x * blockDim.x + threadIdx.x;
  size_t total = (size_t)(E + N) * F;
  if (t >= total) return;
  int f = (int)(t % F);
  size_t e = t / F;
  int hd = f / dout;
  int src, dst;
  if (e < (size_t)E) { src = ei[e]; dst = ei[(size_t)E + e]; }
  else               { src = dst = (int)(e - (size_t)E); }
  float v = h[(size_t)src * F + f] * alpha[e * heads + hd];
  atomicAdd(&out[(size_t)dst * F + f], v);
}

// ---------------- bias + ELU, emit fp32 (last layer) or bf16 (next GEMM) ---
__global__ void k_finalize_elu(const float* __restrict__ acc, const float* __restrict__ bias,
                               int F, size_t total,
                               float* __restrict__ yf32, bf16_t* __restrict__ ybf) {
  size_t i = (size_t)blockIdx.x * blockDim.x + threadIdx.x;
  if (i >= total) return;
  int c = (int)(i % F);
  float v = acc[i] + bias[c];
  v = (v > 0.f) ? v : expm1f(v);
  if (yf32) yf32[i] = v;
  if (ybf)  ybf[i]  = (bf16_t)v;
}

// ---------------- global mean pool + MLP head ------------------------------
__global__ void k_pool(const float* __restrict__ y, const int* __restrict__ batch,
                       float* __restrict__ gsum, float* __restrict__ gcnt, int n) {
  int i = blockIdx.x * blockDim.x + threadIdx.x;
  if (i >= n) return;
  int b = batch[i];
#pragma unroll
  for (int j = 0; j < 16; ++j) atomicAdd(&gsum[b * 16 + j], y[(size_t)i * 16 + j]);
  atomicAdd(&gcnt[b], 1.0f);
}

__global__ void k_mlp_head(const float* __restrict__ gsum, const float* __restrict__ gcnt,
                           const float* __restrict__ fc1w, const float* __restrict__ fc1b,
                           const float* __restrict__ fc2w, const float* __restrict__ fc2b,
                           float* __restrict__ out) {
  int b = threadIdx.x;
  if (b >= 16) return;
  float inv = 1.0f / fmaxf(gcnt[b], 1.0f);
  float g[16];
#pragma unroll
  for (int j = 0; j < 16; ++j) g[j] = gsum[b * 16 + j] * inv;
  float h1[8];
#pragma unroll
  for (int o = 0; o < 8; ++o) {
    float s = fc1b[o];
    for (int j = 0; j < 16; ++j) s += g[j] * fc1w[j * 8 + o];
    h1[o] = fmaxf(s, 0.f);
  }
#pragma unroll
  for (int o = 0; o < 2; ++o) {
    float s = fc2b[o];
    for (int k = 0; k < 8; ++k) s += h1[k] * fc2w[k * 2 + o];
    out[b * 2 + o] = s;
  }
}

// ---------------------------------------------------------------------------

static inline unsigned nblk(size_t n, int t) { return (unsigned)((n + (size_t)t - 1) / (size_t)t); }

extern "C" void kernel_launch(void* const* d_in, const int* in_sizes, int n_in,
                              void* d_out, int out_size, void* d_ws, size_t ws_size,
                              hipStream_t stream) {
  const float* x     = (const float*)d_in[0];
  const int*   ei    = (const int*)  d_in[1];
  const int*   batch = (const int*)  d_in[2];
  const float* W[4]  = {(const float*)d_in[3],  (const float*)d_in[7],
                        (const float*)d_in[11], (const float*)d_in[15]};
  const float* As[4] = {(const float*)d_in[4],  (const float*)d_in[8],
                        (const float*)d_in[12], (const float*)d_in[16]};
  const float* Ad[4] = {(const float*)d_in[5],  (const float*)d_in[9],
                        (const float*)d_in[13], (const float*)d_in[17]};
  const float* Bi[4] = {(const float*)d_in[6],  (const float*)d_in[10],
                        (const float*)d_in[14], (const float*)d_in[18]};
  const float* fc1w = (const float*)d_in[19];
  const float* fc1b = (const float*)d_in[20];
  const float* fc2w = (const float*)d_in[21];
  const float* fc2b = (const float*)d_in[22];
  (void)n_in; (void)out_size; (void)ws_size;

  const int N  = in_sizes[2];        // 20000 nodes
  const int E  = in_sizes[1] / 2;    // 320000 edges
  const int ET = E + N;              // + self loops

  static const int HEADS[4] = {8, 8, 8, 1};
  static const int DOUT[4]  = {128, 64, 32, 16};
  static const int DIN[4]   = {32, 1024, 512, 256};   // layer-0 K padded 4 -> 32

  // ---- workspace carve (256B aligned) ----
  char* ws = (char*)d_ws;
  size_t off = 0;
  auto carve = [&](size_t bytes) -> char* {
    char* p = ws + off;
    off += (bytes + 255) & ~(size_t)255;
    return p;
  };
  bf16_t* actbf = (bf16_t*)carve((size_t)N * 1024 * 2);                 // GEMM input (bf16)
  bf16_t* wbf   = (bf16_t*)carve((size_t)(32*1024 + 1024*512 + 512*256 + 256*16) * 2);
  float*  hbuf  = (float*) carve((size_t)N * 1024 * 4);                 // pre-aggregation h
  float*  obuf  = (float*) carve((size_t)N * 1024 * 4);                 // aggregated output
  float*  als   = (float*) carve((size_t)N * 8 * 4);
  float*  ald   = (float*) carve((size_t)N * 8 * 4);
  float*  emax  = (float*) carve((size_t)N * 8 * 4);
  float*  den   = (float*) carve((size_t)N * 8 * 4);
  float*  ebuf  = (float*) carve((size_t)ET * 8 * 4);                   // scores -> exp -> alpha
  float*  y4    = (float*) carve((size_t)N * 16 * 4);                   // final node features
  float*  gsum  = (float*) carve(16 * 16 * 4);
  float*  gcnt  = (float*) carve(16 * 4);
  float*  tdm_s = (float*) carve(256 * 4);                              // TDM demo sink

  bf16_t* wb[4];
  wb[0] = wbf;
  wb[1] = wb[0] + 32 * 1024;
  wb[2] = wb[1] + 1024 * 512;
  wb[3] = wb[2] + 512 * 256;

  // TDM demo (independent of the real pipeline; writes only to ws scratch)
  k_tdm_copy16x16<<<1, 256, 0, stream>>>(fc1w, tdm_s);

  // ---- convert weights + padded input to bf16 ----
  k_cvt_rows_pad<<<nblk(32 * 1024, 256), 256, 0, stream>>>(W[0], wb[0], 4, 32, 1024);
  k_cvt_rows_pad<<<nblk(1024 * 512, 256), 256, 0, stream>>>(W[1], wb[1], 1024, 1024, 512);
  k_cvt_rows_pad<<<nblk(512 * 256, 256), 256, 0, stream>>>(W[2], wb[2], 512, 512, 256);
  k_cvt_rows_pad<<<nblk(256 * 16, 256), 256, 0, stream>>>(W[3], wb[3], 256, 256, 16);
  k_cvt_cols_pad<<<nblk((size_t)N * 32, 256), 256, 0, stream>>>(x, actbf, N, 4, 32);

  for (int l = 0; l < 4; ++l) {
    const int hs = HEADS[l], dd = DOUT[l], Fo = hs * dd, K = DIN[l];

    // GEMM: h = act @ W  -> hbuf [N, Fo]
    dim3 gb(32, 4);
    dim3 gg((Fo + 63) / 64, (N + 63) / 64);
    k_gemm_bf16<<<gg, gb, 0, stream>>>(actbf, wb[l], hbuf, N, K, Fo);

    // attention logits
    k_node_logits<<<nblk((size_t)N * hs, 256), 256, 0, stream>>>(
        hbuf, As[l], Ad[l], als, ald, N, hs, dd);

    // softmax state + output accumulator init
    k_fill_f32<<<nblk((size_t)N * hs, 256), 256, 0, stream>>>(emax, -INFINITY, (size_t)N * hs);
    k_fill_f32<<<nblk((size_t)N * hs, 256), 256, 0, stream>>>(den, 0.0f, (size_t)N * hs);
    k_fill_f32<<<nblk((size_t)N * Fo, 256), 256, 0, stream>>>(obuf, 0.0f, (size_t)N * Fo);

    const size_t eh = (size_t)ET * hs;
    k_edge_scores<<<nblk(eh, 256), 256, 0, stream>>>(ei, E, N, als, ald, ebuf, emax, hs);
    k_edge_expsum<<<nblk(eh, 256), 256, 0, stream>>>(ei, E, N, ebuf, emax, den, hs);
    k_edge_alpha <<<nblk(eh, 256), 256, 0, stream>>>(ei, E, N, ebuf, den, hs);

    const size_t ef = (size_t)ET * Fo;
    k_edge_aggregate<<<nblk(ef, 256), 256, 0, stream>>>(ei, E, N, hbuf, ebuf, obuf, hs, dd);

    const size_t nf = (size_t)N * Fo;
    k_finalize_elu<<<nblk(nf, 256), 256, 0, stream>>>(
        obuf, Bi[l], Fo, nf,
        (l == 3) ? y4 : (float*)nullptr,
        (l < 3) ? actbf : (bf16_t*)nullptr);
  }

  // global mean pool + MLP head
  k_fill_f32<<<1, 256, 0, stream>>>(gsum, 0.0f, (size_t)(16 * 16));
  k_fill_f32<<<1, 32, 0, stream>>>(gcnt, 0.0f, (size_t)16);
  k_pool<<<nblk((size_t)N, 256), 256, 0, stream>>>(y4, batch, gsum, gcnt, N);
  k_mlp_head<<<1, 16, 0, stream>>>(gsum, gcnt, fc1w, fc1b, fc2w, fc2b, (float*)d_out);
}
